// TransformerEncoder_13932873908290
// MI455X (gfx1250) — matrix operand order repaired and checked by reading, not compiled, not run on hardware
//
#include <hip/hip_runtime.h>
#include <hip/hip_bf16.h>
#include <stdint.h>

// ---------------------------------------------------------------------------
// Types for CDNA5 WMMA bf16 path
// ---------------------------------------------------------------------------
typedef __bf16 bf16_t;
typedef __attribute__((ext_vector_type(16))) __bf16    v16bf;
typedef __attribute__((ext_vector_type(8)))  __bf16    v8bf;
typedef __attribute__((ext_vector_type(8)))  float     v8f;
typedef __attribute__((ext_vector_type(4)))  uint32_t  u32x4;
typedef __attribute__((ext_vector_type(8)))  uint32_t  u32x8;

#define DIM   1024
#define SEQ   2048
#define BATCH 4
#define ROWS  (SEQ * BATCH)   // 8192

// f32 -> bf16, round-to-nearest-even
static __device__ __forceinline__ bf16_t f2bf(float f) {
    union { float f; uint32_t u; } v; v.f = f;
    uint32_t r = v.u + 0x7FFFu + ((v.u >> 16) & 1u);
    uint16_t h = (uint16_t)(r >> 16);
    return __builtin_bit_cast(bf16_t, h);
}

static __device__ __forceinline__ v16bf join16(v8bf lo, v8bf hi) {
    return __builtin_shufflevector(lo, hi, 0,1,2,3,4,5,6,7,8,9,10,11,12,13,14,15);
}

// LDS byte address of a shared-memory pointer (generic addr low 32 bits).
static __device__ __forceinline__ uint32_t lds_addr_of(const void* p) {
    return (uint32_t)(uintptr_t)p;
}

// ---------------------------------------------------------------------------
// Tensor Data Mover: 2D tile load Global->LDS (D# per ISA 08_async_tensor §8).
// data_size = 2 bytes; padding encodes bank-friendly, 16B-aligned LDS strides.
// Issued by one wave; completion via TENSORcnt (in-order per wave).
// ---------------------------------------------------------------------------
static __device__ __forceinline__ void tdm_load_2d(
    uint32_t lds_addr, const void* gaddr,
    uint32_t tensor_d0, uint32_t tensor_d1,       // tensor extents (elems)
    uint32_t tile_d0,   uint32_t tile_d1,         // tile extents   (elems)
    uint32_t stride0,                              // dim0 stride    (elems)
    uint32_t pad_interval_code, uint32_t pad_amount_val)
{
    uint64_t ga = (uint64_t)gaddr;
    u32x4 g0;
    g0.x = 1u;                                     // count=1, user descriptor
    g0.y = lds_addr;                               // lds_addr [63:32]
    g0.z = (uint32_t)ga;                           // global_addr [95:64]
    g0.w = (uint32_t)((ga >> 32) & 0x1FFFFFFu)     // global_addr [120:96]
         | (2u << 30);                             // type=2 ("image") [127:126]
    u32x8 g1;
    g1.s0 = (1u << 16)                             // data_size = 2 bytes
          | (1u << 20)                             // pad_enable
          | (pad_interval_code << 22)
          | (pad_amount_val    << 25);
    g1.s1 = (tensor_d0 & 0xFFFFu) << 16;           // tensor_dim0 lo -> [79:48]
    g1.s2 = ((tensor_d0 >> 16) & 0xFFFFu)          // tensor_dim0 hi
          | ((tensor_d1 & 0xFFFFu) << 16);         // tensor_dim1 lo
    g1.s3 = ((tensor_d1 >> 16) & 0xFFFFu)          // tensor_dim1 hi
          | (tile_d0 << 16);                       // tile_dim0 [127:112]
    g1.s4 = tile_d1 & 0xFFFFu;                     // tile_dim1 [143:128], tile_dim2=0
    g1.s5 = stride0;                               // tensor_dim0_stride lo32
    g1.s6 = 0u;
    g1.s7 = 0u;
    u32x4 gz = {0u, 0u, 0u, 0u};
    asm volatile("tensor_load_to_lds %0, %1, %2, %3"
                 :: "s"(g0), "s"(g1), "s"(gz), "s"(gz)
                 : "memory");
}

// ---------------------------------------------------------------------------
// Pre-conversion kernels (one-shot, bandwidth-bound)
// ---------------------------------------------------------------------------
__global__ __launch_bounds__(256)
void conv_f32_bf16(const float* __restrict__ in, bf16_t* __restrict__ out)
{
    size_t i = (size_t)blockIdx.x * 256 + threadIdx.x;     // one float4 each
    float4 f = ((const float4*)in)[i];
    union { bf16_t h[4]; uint2 u; } pk;
    pk.h[0] = f2bf(f.x); pk.h[1] = f2bf(f.y);
    pk.h[2] = f2bf(f.z); pk.h[3] = f2bf(f.w);
    ((uint2*)out)[i] = pk.u;
}

// Wt[k][n] = bf16(W[n][k])  (1024x1024)
__global__ __launch_bounds__(256)
void transpose_conv(const float* __restrict__ W, bf16_t* __restrict__ Wt)
{
    __shared__ float t[16][17];
    int tx = threadIdx.x, ty = threadIdx.y;
    int nb = blockIdx.x * 16, kb = blockIdx.y * 16;
    t[ty][tx] = W[(size_t)(nb + ty) * DIM + kb + tx];
    __syncthreads();
    Wt[(size_t)(kb + ty) * DIM + nb + tx] = f2bf(t[tx][ty]);
}

// ---------------------------------------------------------------------------
// GEMM: C[m,n] = sum_k A[m,k]*Wt[k,n] + bias[n]  (A bf16 row-major,
// Wt bf16 k-major).  Block 256 (8 waves), tile 128x64, wave 32x32 (2x2 WMMA).
// Double-buffered TDM staging: tile i+1 streams while tile i is computed.
//   A rows 64B  + 16B pad => ASTR 40 elems
//   B rows 128B + 16B pad => BSTR 72 elems
// OMODE: 0 = f32 row-major, 1 = bf16 row-major, 2 = bf16 Kt[b][n][s]
// ---------------------------------------------------------------------------
#define ASTR 40
#define BSTR 72
#define KSTEPS (DIM / 32)   // 32

template <int OMODE>
__global__ __launch_bounds__(256)
void gemm_bf16_tdm(const bf16_t* __restrict__ A, const bf16_t* __restrict__ Bt,
                   const float* __restrict__ bias, void* __restrict__ Cout)
{
    __shared__ bf16_t As[2][128 * ASTR];
    __shared__ bf16_t Bs[2][32 * BSTR];

    const int tid  = threadIdx.x;
    const int lane = tid & 31;
    const int wave = tid >> 5;      // 0..7
    const int wr   = wave & 3;      // 4 row groups of 32
    const int wc   = wave >> 2;     // 2 col groups of 32
    const int half = lane >> 4;
    const int l15  = lane & 15;
    const int mBase = blockIdx.x * 128;
    const int nBase = blockIdx.y * 64;

    v8f acc[2][2] = {};

    // Prologue: stage tile 0 into buffer 0
    if (wave == 0) {
        tdm_load_2d(lds_addr_of(As[0]), A + (size_t)mBase * DIM,
                    DIM, ROWS, 32, 128, DIM, /*16DW*/3, /*4DW*/3);
        tdm_load_2d(lds_addr_of(Bs[0]), Bt + nBase,
                    DIM, DIM, 64, 32, DIM, /*32DW*/4, /*4DW*/3);
    }

    for (int it = 0; it < KSTEPS; ++it) {
        __syncthreads();    // prior compute on buffer (it+1)&1 has finished
        if (wave == 0) {
            if (it + 1 < KSTEPS) {
                const int kc = (it + 1) * 32;
                tdm_load_2d(lds_addr_of(As[(it + 1) & 1]),
                            A + (size_t)mBase * DIM + kc,
                            DIM, ROWS, 32, 128, DIM, 3, 3);
                tdm_load_2d(lds_addr_of(Bs[(it + 1) & 1]),
                            Bt + (size_t)kc * DIM + nBase,
                            DIM, DIM, 64, 32, DIM, 4, 3);
                // 2 newest in flight; 2 oldest (tile it) complete (in-order)
                __builtin_amdgcn_s_wait_tensorcnt(2);
            } else {
                __builtin_amdgcn_s_wait_tensorcnt(0);
            }
        }
        __syncthreads();    // tile it visible to all waves

        const bf16_t* Asb = As[it & 1];
        const bf16_t* Bsb = Bs[it & 1];
        v16bf af[2], bf[2];
        #pragma unroll
        for (int i = 0; i < 2; ++i) {
            const bf16_t* p = &Asb[(wr * 32 + i * 16 + l15) * ASTR + half * 8];
            af[i] = join16(*(const v8bf*)p, *(const v8bf*)(p + 16));
        }
        #pragma unroll
        for (int j = 0; j < 2; ++j) {
            const bf16_t* p = &Bsb[half * 8 * BSTR + wc * 32 + j * 16 + l15];
            #pragma unroll
            for (int s = 0; s < 16; ++s) bf[j][s] = p[(s + (s & 8)) * BSTR];
        }
        #pragma unroll
        for (int i = 0; i < 2; ++i)
            #pragma unroll
            for (int j = 0; j < 2; ++j)
                acc[i][j] = __builtin_amdgcn_wmma_f32_16x16x32_bf16(
                    false, af[i], false, bf[j], (short)0, acc[i][j], false, false);
    }

    // Epilogue. C layout: VGPR v -> row v + 8*half, col = lane&15
    #pragma unroll
    for (int i = 0; i < 2; ++i) {
        #pragma unroll
        for (int j = 0; j < 2; ++j) {
            int n = nBase + wc * 32 + j * 16 + l15;
            float bv = bias[n];
            #pragma unroll
            for (int v = 0; v < 8; ++v) {
                int m = mBase + wr * 32 + i * 16 + v + half * 8;
                float r = acc[i][j][v] + bv;
                if (OMODE == 0) {
                    ((float*)Cout)[(size_t)m * DIM + n] = r;
                } else if (OMODE == 1) {
                    ((bf16_t*)Cout)[(size_t)m * DIM + n] = f2bf(r);
                } else {  // Kt[b][n][s], b = m&3, s = m>>2
                    int b = m & (BATCH - 1), s = m >> 2;
                    ((bf16_t*)Cout)[((size_t)b * DIM + n) * SEQ + s] = f2bf(r);
                }
            }
        }
    }
}

// ---------------------------------------------------------------------------
// Fused scores + running argmax (never materializes [B,Sq,Sk]).
// grid = (S/64, B, KSEG); block 128 (4 waves x 16 q-rows).
// Q tile (64x1024 bf16) staged once by a strided TDM (row stride B*D);
// K tiles double-buffered from the pre-transposed Kt[b][d][s].
//   Q rows 2048B + 16B pad per 1024B => QSTR 1040 elems (16B-aligned skew)
//   K rows 32B   + 16B pad           => KSTR 24 elems
// ---------------------------------------------------------------------------
#define QSTR 1040
#define KSTR 24
#define KSEG 4
#define KT_PER_SEG (SEQ / 16 / KSEG)   // 32

__global__ __launch_bounds__(128)
void scores_argmax(const bf16_t* __restrict__ Qb, const bf16_t* __restrict__ Kt,
                   float* __restrict__ MX, int* __restrict__ IX)
{
    extern __shared__ __align__(16) char smem[];
    bf16_t* Qs  = (bf16_t*)smem;                 // [64][QSTR]
    bf16_t* Ks0 = Qs + 64 * QSTR;                // [1024][KSTR] x2 buffers
    bf16_t* Ks1 = Ks0 + DIM * KSTR;

    const int tid  = threadIdx.x;        // 0..127
    const int lane = tid & 31;
    const int wave = tid >> 5;           // 0..3
    const int half = lane >> 4;
    const int l15  = lane & 15;
    const int qbase = blockIdx.x * 64;
    const int b     = blockIdx.y;
    const int seg   = blockIdx.z;
    const bf16_t* kplane = Kt + (size_t)b * DIM * SEQ;

    if (wave == 0) {
        // Q: 64 q-rows x 1024 d, row stride B*D elems handles [S,B,D] layout
        tdm_load_2d(lds_addr_of(Qs), Qb + ((size_t)qbase * BATCH + b) * DIM,
                    DIM, ROWS, DIM, 64, BATCH * DIM, /*256DW*/7, /*4DW*/3);
        // K tile 0: 1024 d-rows x 16 s-cols from Kt[b][d][s]
        tdm_load_2d(lds_addr_of(Ks0), kplane + (size_t)seg * KT_PER_SEG * 16,
                    SEQ, DIM, 16, DIM, SEQ, /*8DW*/2, /*4DW*/3);
    }

    float rmax[8]; int ridx[8];
    #pragma unroll
    for (int v = 0; v < 8; ++v) { rmax[v] = -__builtin_inff(); ridx[v] = 0; }

    for (int t = 0; t < KT_PER_SEG; ++t) {
        const int kt = seg * KT_PER_SEG + t;
        __syncthreads();    // prior compute on the other K buffer finished
        if (wave == 0) {
            if (t + 1 < KT_PER_SEG) {
                bf16_t* nxt = ((t + 1) & 1) ? Ks1 : Ks0;
                tdm_load_2d(lds_addr_of(nxt), kplane + (size_t)(kt + 1) * 16,
                            SEQ, DIM, 16, DIM, SEQ, 2, 3);
                // newest K tile in flight; Q + K tile t complete (in-order)
                __builtin_amdgcn_s_wait_tensorcnt(1);
            } else {
                __builtin_amdgcn_s_wait_tensorcnt(0);
            }
        }
        __syncthreads();

        const bf16_t* Ksb = (t & 1) ? Ks1 : Ks0;
        v8f acc = {};
        for (int dc = 0; dc < DIM; dc += 32) {
            const int qoff = dc + ((dc >> 9) << 3);   // +8 elems per 512 (16B pads)
            const bf16_t* pa = &Qs[(wave * 16 + l15) * QSTR + qoff + half * 8];
            v16bf a = join16(*(const v8bf*)pa, *(const v8bf*)(pa + 16));
            const bf16_t* pb = &Ksb[(dc + half * 8) * KSTR + l15];   // B col n == l15, k in table order
            v16bf bb;
            #pragma unroll
            for (int s = 0; s < 16; ++s) bb[s] = pb[(s + (s & 8)) * KSTR];
            acc = __builtin_amdgcn_wmma_f32_16x16x32_bf16(
                false, a, false, bb, (short)0, acc, false, false);
        }
        const int col = kt * 16 + l15;
        #pragma unroll
        for (int v = 0; v < 8; ++v) {
            float s = acc[v];
            if (s > rmax[v]) { rmax[v] = s; ridx[v] = col; }
        }
    }

    // Combine (max, argmax) across the 16 lanes of each N group.
    #pragma unroll
    for (int v = 0; v < 8; ++v) {
        #pragma unroll
        for (int off = 1; off < 16; off <<= 1) {
            float om = __shfl_xor(rmax[v], off, 32);
            int   oi = __shfl_xor(ridx[v], off, 32);
            if (om > rmax[v] || (om == rmax[v] && oi < ridx[v])) { rmax[v] = om; ridx[v] = oi; }
        }
        if (l15 == 0) {
            int q = qbase + wave * 16 + v + half * 8;
            MX[(size_t)seg * ROWS + b * SEQ + q] = rmax[v];
            IX[(size_t)seg * ROWS + b * SEQ + q] = ridx[v];
        }
    }
}

// ---------------------------------------------------------------------------
// Merge KSEG candidates; gather V row (f32) -> P (bf16), or zeros.
// One block per output row (s*B+b).
// ---------------------------------------------------------------------------
__global__ __launch_bounds__(256)
void gather_v(const float* __restrict__ V, const float* __restrict__ MX,
              const int* __restrict__ IX, bf16_t* __restrict__ P)
{
    const int bid = blockIdx.x;          // s*BATCH + b
    const int b = bid & (BATCH - 1);
    const int s = bid >> 2;
    const size_t off = (size_t)b * SEQ + s;

    float m = -__builtin_inff(); int k = 0;
    #pragma unroll
    for (int seg = 0; seg < KSEG; ++seg) {
        float ms = MX[(size_t)seg * ROWS + off];
        int   ks = IX[(size_t)seg * ROWS + off];
        if (ms > m || (ms == m && ks < k)) { m = ms; k = ks; }
    }

    float4 f = make_float4(0.f, 0.f, 0.f, 0.f);
    if (m >= 0.95f) {
        f = ((const float4*)(V + ((size_t)k * BATCH + b) * DIM))[threadIdx.x];
    }
    union { bf16_t h[4]; uint2 u; } pk;
    pk.h[0] = f2bf(f.x); pk.h[1] = f2bf(f.y);
    pk.h[2] = f2bf(f.z); pk.h[3] = f2bf(f.w);
    ((uint2*)(P + (size_t)bid * DIM))[threadIdx.x] = pk.u;
}

// ---------------------------------------------------------------------------
// Host-side launcher
// ---------------------------------------------------------------------------
extern "C" void kernel_launch(void* const* d_in, const int* in_sizes, int n_in,
                              void* d_out, int out_size, void* d_ws, size_t ws_size,
                              hipStream_t stream) {
    (void)in_sizes; (void)n_in; (void)out_size; (void)ws_size;
    const float* query = (const float*)d_in[0];
    const float* key   = (const float*)d_in[1];
    const float* value = (const float*)d_in[2];
    const float* Wq = (const float*)d_in[3];
    const float* Wk = (const float*)d_in[4];
    const float* Wv = (const float*)d_in[5];
    const float* bq = (const float*)d_in[6];
    const float* bk = (const float*)d_in[7];
    const float* bv = (const float*)d_in[8];
    const float* Wo = (const float*)d_in[9];
    const float* bo = (const float*)d_in[10];
    float* out = (float*)d_out;

    // Workspace layout
    char* w = (char*)d_ws;
    bf16_t* Qa  = (bf16_t*)w; w += (size_t)ROWS * DIM * 2;   // bf16(query)
    bf16_t* Ka  = (bf16_t*)w; w += (size_t)ROWS * DIM * 2;   // bf16(key)
    bf16_t* Va  = (bf16_t*)w; w += (size_t)ROWS * DIM * 2;   // bf16(value)
    bf16_t* Wqt = (bf16_t*)w; w += (size_t)DIM * DIM * 2;
    bf16_t* Wkt = (bf16_t*)w; w += (size_t)DIM * DIM * 2;
    bf16_t* Wvt = (bf16_t*)w; w += (size_t)DIM * DIM * 2;
    bf16_t* Wot = (bf16_t*)w; w += (size_t)DIM * DIM * 2;
    bf16_t* Qp  = (bf16_t*)w; w += (size_t)ROWS * DIM * 2;   // Q proj, bf16 [S,B,D]
    bf16_t* Ktp = (bf16_t*)w; w += (size_t)BATCH * DIM * SEQ * 2; // K proj, [b][d][s]
    float*  Vf  = (float*)w;  w += (size_t)ROWS * DIM * 4;   // V proj, f32 [S,B,D]
    bf16_t* P   = (bf16_t*)w; w += (size_t)ROWS * DIM * 2;   // gathered rows, bf16
    float*  MX  = (float*)w;  w += (size_t)KSEG * ROWS * 4;
    int*    IX  = (int*)w;

    const int convBlocks = (ROWS * DIM / 4) / 256;           // float4 per thread
    conv_f32_bf16<<<convBlocks, 256, 0, stream>>>(query, Qa);
    conv_f32_bf16<<<convBlocks, 256, 0, stream>>>(key,   Ka);
    conv_f32_bf16<<<convBlocks, 256, 0, stream>>>(value, Va);

    dim3 tGrid(DIM / 16, DIM / 16), tBlk(16, 16);
    transpose_conv<<<tGrid, tBlk, 0, stream>>>(Wq, Wqt);
    transpose_conv<<<tGrid, tBlk, 0, stream>>>(Wk, Wkt);
    transpose_conv<<<tGrid, tBlk, 0, stream>>>(Wv, Wvt);
    transpose_conv<<<tGrid, tBlk, 0, stream>>>(Wo, Wot);

    dim3 gemmGrid(ROWS / 128, DIM / 64);
    gemm_bf16_tdm<1><<<gemmGrid, 256, 0, stream>>>(Qa, Wqt, bq, Qp);
    gemm_bf16_tdm<2><<<gemmGrid, 256, 0, stream>>>(Ka, Wkt, bk, Ktp);
    gemm_bf16_tdm<0><<<gemmGrid, 256, 0, stream>>>(Va, Wvt, bv, Vf);

    size_t shmem = (size_t)(64 * QSTR + 2 * DIM * KSTR) * sizeof(bf16_t); // ~231 KB
    scores_argmax<<<dim3(SEQ / 64, BATCH, KSEG), 128, shmem, stream>>>(Qp, Ktp, MX, IX);

    gather_v<<<dim3(ROWS), 256, 0, stream>>>(Vf, MX, IX, P);

    gemm_bf16_tdm<0><<<gemmGrid, 256, 0, stream>>>(P, Wot, bo, out);
}
